// CommunityGNNEncoder_59785944760475
// MI455X (gfx1250) — matrix operand order repaired and checked by reading, not compiled, not run on hardware
//
#include <hip/hip_runtime.h>
#include <stdint.h>

#define D_DIM 128
#define NEG_SLOPE 0.2f
#define EPS_DENOM 1e-16f

typedef __attribute__((ext_vector_type(16))) _Float16 v16h;
typedef __attribute__((ext_vector_type(8)))  _Float16 v8h;
typedef __attribute__((ext_vector_type(4)))  _Float16 v4h;
typedef __attribute__((ext_vector_type(8)))  float    v8f;

// Order-preserving float<->uint key so we can use integer atomicMax for the
// segment max (works for negative logits, unlike raw-bits compare).
__device__ __forceinline__ unsigned fkey(float f) {
  unsigned u = __float_as_uint(f);
  return (u & 0x80000000u) ? ~u : (u | 0x80000000u);
}
__device__ __forceinline__ float funkey(unsigned k) {
  unsigned u = (k & 0x80000000u) ? (k ^ 0x80000000u) : ~k;
  return __uint_as_float(u);
}

// ---------------------------------------------------------------------------
// v_src = W_src @ att_src, v_dst = W_dst @ att_dst  (two 128-vectors).
// This removes the need for the full xd = x @ W_dst GEMM entirely.
__global__ void k_att_vec(const float* __restrict__ Wsrc, const float* __restrict__ Wdst,
                          const float* __restrict__ as, const float* __restrict__ ad,
                          float* __restrict__ vsrc, float* __restrict__ vdst) {
  int t = threadIdx.x;
  if (t < 128) {
    float acc = 0.f;
    for (int h = 0; h < 128; ++h) acc += Wsrc[t * 128 + h] * as[h];
    vsrc[t] = acc;
  } else if (t < 256) {
    int k = t - 128;
    float acc = 0.f;
    for (int h = 0; h < 128; ++h) acc += Wdst[k * 128 + h] * ad[h];
    vdst[k] = acc;
  }
}

// Transpose + f16-convert weights so WMMA B-fragments are contiguous 32B loads.
__global__ void k_wprep(const float* __restrict__ Wsrc, const float* __restrict__ Wlin,
                        _Float16* __restrict__ WsrcT, _Float16* __restrict__ WlinT) {
  int t = blockIdx.x * blockDim.x + threadIdx.x;
  if (t < 128 * 128) {
    int n = t & 127;                 // Wsrc is [K=128][N=128] row major
    int k = t >> 7;
    WsrcT[n * 128 + k] = (_Float16)Wsrc[t];
  } else if (t < 128 * 128 + 128 * 64) {
    int i = t - 128 * 128;           // Wlin is [K=128][N=64] row major
    int k = i >> 6;
    int n = i & 63;
    WlinT[n * 128 + k] = (_Float16)Wlin[i];
  }
}

// Per node (one wave each): convert x row to f16, and compute
// a_src[n] = x[n].v_src, a_dst[n] = x[n].v_dst via lane-parallel dot + shfl.
__global__ void k_node_prep(const float* __restrict__ x, const float* __restrict__ vsrc,
                            const float* __restrict__ vdst, _Float16* __restrict__ xh,
                            float* __restrict__ a_src, float* __restrict__ a_dst, int N) {
  const int node = blockIdx.x * (blockDim.x >> 5) + (threadIdx.x >> 5);
  const int lane = threadIdx.x & 31;
  if (node >= N) return;
  const float4 xv = ((const float4*)(x + (size_t)node * D_DIM))[lane];
  v4h hv = { (_Float16)xv.x, (_Float16)xv.y, (_Float16)xv.z, (_Float16)xv.w };
  ((v4h*)(xh + (size_t)node * D_DIM))[lane] = hv;
  const float4 vs = ((const float4*)vsrc)[lane];
  const float4 vd = ((const float4*)vdst)[lane];
  float s = xv.x * vs.x + xv.y * vs.y + xv.z * vs.z + xv.w * vs.w;
  float d = xv.x * vd.x + xv.y * vd.y + xv.z * vd.z + xv.w * vd.w;
#pragma unroll
  for (int off = 16; off; off >>= 1) {
    s += __shfl_xor(s, off, 32);
    d += __shfl_xor(d, off, 32);
  }
  if (lane == 0) { a_src[node] = s; a_dst[node] = d; }
}

// Zero agg/denom, init max-keys to key(-inf).
__global__ void k_init(float* __restrict__ agg, float* __restrict__ denom,
                       unsigned* __restrict__ mkey, int N) {
  int t = blockIdx.x * blockDim.x + threadIdx.x;
  if (t < N * D_DIM) agg[t] = 0.f;
  if (t < N) { denom[t] = 0.f; mkey[t] = 0x007FFFFFu; /* fkey(-inf) */ }
}

// Edge pass 1: logits (leaky relu) + segment max via integer atomicMax.
// Edges i < E come from edge_index; i >= E are the implicit self loops.
__global__ void k_edge_logit(const int* __restrict__ ei, const float* __restrict__ a_src,
                             const float* __restrict__ a_dst, float* __restrict__ ebuf,
                             unsigned* __restrict__ mkey, int E, int N) {
  int i = blockIdx.x * blockDim.x + threadIdx.x;
  if (i >= E + N) return;
  int s, d;
  if (i < E) { s = ei[i]; d = ei[E + i]; } else { s = d = i - E; }
  float e = a_src[s] + a_dst[d];
  e = (e > 0.f) ? e : NEG_SLOPE * e;
  ebuf[i] = e;
  atomicMax(mkey + d, fkey(e));
}

// Edge pass 2: ex = exp(e - m[dst]); denom[dst] += ex.
__global__ void k_edge_exp(const int* __restrict__ ei, const unsigned* __restrict__ mkey,
                           float* __restrict__ ebuf, float* __restrict__ denom, int E, int N) {
  int i = blockIdx.x * blockDim.x + threadIdx.x;
  if (i >= E + N) return;
  int d = (i < E) ? ei[E + i] : (i - E);
  float m = funkey(mkey[d]);
  float v = __expf(ebuf[i] - m);
  ebuf[i] = v;
  atomicAdd(denom + d, v);
}

// Edge pass 3 (one wave per edge): agg[dst] += alpha * xs[src].
// float4 gather of xs row (coalesced 512B per wave); xs+agg are L2-resident.
__global__ void k_edge_agg(const int* __restrict__ ei, const float* __restrict__ ebuf,
                           const float* __restrict__ denom, const float* __restrict__ xs,
                           float* __restrict__ agg, int E, int N) {
  const int e = blockIdx.x * (blockDim.x >> 5) + (threadIdx.x >> 5);
  const int lane = threadIdx.x & 31;
  if (e >= E + N) return;
  int s, d;
  if (e < E) { s = ei[e]; d = ei[E + e]; } else { s = d = e - E; }
  const float alpha = ebuf[e] / (denom[d] + EPS_DENOM);
  const float4 xv = ((const float4*)(xs + (size_t)s * D_DIM))[lane];
  float* ag = agg + (size_t)d * D_DIM + lane * 4;
  atomicAdd(ag + 0, alpha * xv.x);
  atomicAdd(ag + 1, alpha * xv.y);
  atomicAdd(ag + 2, alpha * xv.z);
  atomicAdd(ag + 3, alpha * xv.w);
}

// h = relu(agg + bias_conv), converted to f16 for the final WMMA GEMM.
__global__ void k_hprep(const float* __restrict__ agg, const float* __restrict__ bias,
                        _Float16* __restrict__ hh, int N) {
  int t = blockIdx.x * blockDim.x + threadIdx.x;
  if (t >= N * D_DIM) return;
  float v = agg[t] + bias[t & 127];
  hh[t] = (_Float16)fmaxf(v, 0.f);
}

// ---------------------------------------------------------------------------
// WMMA GEMM: C[nrows x ncols] = A[nrows x 128] * B[128 x ncols] (+bias).
// A is f16 row-major; BT is B transposed f16 row-major (ncols x 128).
// One wave computes a 16x64 tile: A-fragment loaded once per K-chunk (two
// contiguous v8h loads giving K = 16*lhi + e, same order as B), 4 B-fragments
// (one v16h = 32B contiguous load each), 4 f32 accumulators.
__global__ void k_gemm16x64(const _Float16* __restrict__ A, const _Float16* __restrict__ BT,
                            const float* __restrict__ bias, float* __restrict__ C,
                            int nrows, int ncols) {
  const int wid = blockIdx.x * (blockDim.x >> 5) + (threadIdx.x >> 5);
  const int ngrp = ncols >> 6;                  // 64-wide column groups
  const int mtiles = (nrows + 15) >> 4;
  if (wid >= mtiles * ngrp) return;
  const int m0 = (wid / ngrp) << 4;
  const int n0 = (wid % ngrp) << 6;
  const int lane = threadIdx.x & 31;
  const int lrow = lane & 15;
  const int lhi  = lane >> 4;
  int arow = m0 + lrow;
  if (arow >= nrows) arow = nrows - 1;          // clamp; tail stores guarded
  const _Float16* ar  = A + (size_t)arow * D_DIM;
  const _Float16* b0p = BT + (size_t)(n0 + lrow) * D_DIM;
  v8f c0 = {}, c1 = {}, c2 = {}, c3 = {};
#pragma unroll
  for (int k0 = 0; k0 < D_DIM; k0 += 32) {
    // A 16x32 f16 fragment: lanes 0-15 rows, elems 0-15 -> K=k0+16*lhi+e
    // (same K order as B).
    v8h alo = *(const v8h*)(ar + k0 + lhi * 16);
    v8h ahi = *(const v8h*)(ar + k0 + lhi * 16 + 8);
    v16h a = __builtin_shufflevector(alo, ahi, 0, 1, 2, 3, 4, 5, 6, 7,
                                     8, 9, 10, 11, 12, 13, 14, 15);
    // B 32x16 fragment: lane holds col n0+lrow, K = k0 + 16*lhi + e.
    const int kb = k0 + lhi * 16;
    v16h b0 = *(const v16h*)(b0p + kb);
    v16h b1 = *(const v16h*)(b0p + 16 * D_DIM + kb);
    v16h b2 = *(const v16h*)(b0p + 32 * D_DIM + kb);
    v16h b3 = *(const v16h*)(b0p + 48 * D_DIM + kb);
    c0 = __builtin_amdgcn_wmma_f32_16x16x32_f16(false, a, false, b0, (short)0, c0, false, false);
    c1 = __builtin_amdgcn_wmma_f32_16x16x32_f16(false, a, false, b1, (short)0, c1, false, false);
    c2 = __builtin_amdgcn_wmma_f32_16x16x32_f16(false, a, false, b2, (short)0, c2, false, false);
    c3 = __builtin_amdgcn_wmma_f32_16x16x32_f16(false, a, false, b3, (short)0, c3, false, false);
  }
  // D layout: VGPR v, lane L -> row = 8*(L>>4)+v, col = L&15.
#pragma unroll
  for (int v = 0; v < 8; ++v) {
    const int row = m0 + lhi * 8 + v;
    if (row < nrows) {
      float* cr = C + (size_t)row * ncols;
      const int cA = n0 + lrow;
      float bA = 0.f, bB = 0.f, bC = 0.f, bD = 0.f;
      if (bias) { bA = bias[cA]; bB = bias[cA + 16]; bC = bias[cA + 32]; bD = bias[cA + 48]; }
      cr[cA]      = c0[v] + bA;
      cr[cA + 16] = c1[v] + bB;
      cr[cA + 32] = c2[v] + bC;
      cr[cA + 48] = c3[v] + bD;
    }
  }
}

// ---------------------------------------------------------------------------
extern "C" void kernel_launch(void* const* d_in, const int* in_sizes, int n_in,
                              void* d_out, int out_size, void* d_ws, size_t ws_size,
                              hipStream_t stream) {
  const float* x         = (const float*)d_in[0];
  const float* Wsrc      = (const float*)d_in[1];
  const float* Wdst      = (const float*)d_in[2];
  const float* att_s     = (const float*)d_in[3];
  const float* att_d     = (const float*)d_in[4];
  const float* bias_conv = (const float*)d_in[5];
  const float* Wlin      = (const float*)d_in[6];
  const float* blin      = (const float*)d_in[7];
  const int*   ei        = (const int*)d_in[8];

  const int N = in_sizes[0] / D_DIM;   // 100000
  const int E = in_sizes[8] / 2;       // 1600000
  const int EN = E + N;                // edges + self loops

  // Workspace carve-up (256B aligned slices).
  char* p = (char*)d_ws;
  auto carve = [&](size_t bytes) -> char* {
    char* r = p;
    p += (bytes + 255) & ~(size_t)255;
    return r;
  };
  _Float16* xh    = (_Float16*)carve((size_t)N * D_DIM * 2);
  float*    xs    = (float*)carve((size_t)N * D_DIM * 4);
  float*    agg   = (float*)carve((size_t)N * D_DIM * 4);
  float*    a_src = (float*)carve((size_t)N * 4);
  float*    a_dst = (float*)carve((size_t)N * 4);
  unsigned* mkey  = (unsigned*)carve((size_t)N * 4);
  float*    denom = (float*)carve((size_t)N * 4);
  float*    ebuf  = (float*)carve((size_t)EN * 4);
  float*    vsrc  = (float*)carve(512);
  float*    vdst  = (float*)carve(512);
  _Float16* WsrcT = (_Float16*)carve(128 * 128 * 2);
  _Float16* WlinT = (_Float16*)carve(64 * 128 * 2);
  _Float16* hh    = xh;                // reuse: xh dead after the first GEMM

  const int mtiles = (N + 15) / 16;

  k_init<<<(N * D_DIM + 255) / 256, 256, 0, stream>>>(agg, denom, mkey, N);
  k_att_vec<<<1, 256, 0, stream>>>(Wsrc, Wdst, att_s, att_d, vsrc, vdst);
  k_wprep<<<(128 * 128 + 128 * 64 + 255) / 256, 256, 0, stream>>>(Wsrc, Wlin, WsrcT, WlinT);
  k_node_prep<<<(N + 7) / 8, 256, 0, stream>>>(x, vsrc, vdst, xh, a_src, a_dst, N);

  // xs = x @ W_src  (f16 WMMA, f32 accumulate)
  {
    int waves = mtiles * 2;            // ncols=128 -> two 64-wide groups
    k_gemm16x64<<<(waves + 7) / 8, 256, 0, stream>>>(xh, WsrcT, nullptr, xs, N, 128);
  }

  k_edge_logit<<<(EN + 255) / 256, 256, 0, stream>>>(ei, a_src, a_dst, ebuf, mkey, E, N);
  k_edge_exp<<<(EN + 255) / 256, 256, 0, stream>>>(ei, mkey, ebuf, denom, E, N);
  k_edge_agg<<<(EN + 7) / 8, 256, 0, stream>>>(ei, ebuf, denom, xs, agg, E, N);

  k_hprep<<<(N * D_DIM + 255) / 256, 256, 0, stream>>>(agg, bias_conv, hh, N);

  // out = relu(agg + bias) @ W_lin + b_lin
  k_gemm16x64<<<(mtiles + 7) / 8, 256, 0, stream>>>(hh, WlinT, blin, (float*)d_out, N, 64);
}
